// BERT_CRF_NER_28656021799093
// MI455X (gfx1250) — compile-verified, matrix operand order checked by reading, not run on hardware
//
#include <hip/hip_runtime.h>
#include <hip/hip_bf16.h>

// Problem constants (from reference): B=128, S=512, H=768, T=20
#define BSZ 128
#define SSZ 512
#define HSZ 768
#define TSZ 20
#define TPAD 32            // padded tag stride for feats workspace
#define START_TAG 18
#define STOP_TAG 19
#define KCHUNKS (HSZ / 32) // 24 K-chunks of 32

typedef __attribute__((ext_vector_type(16))) _Float16 v16h;
typedef __attribute__((ext_vector_type(8)))  float    v8f;

// Load a 16-element f16 A-fragment for one lane:
// K values {p[0..7]} and {p[16..23]}  (per CDNA5 16-bit A-matrix 16x32 layout,
// caller pre-offsets p by the lane's K-half).
__device__ __forceinline__ v16h load_frag16(const float* __restrict__ p) {
    v16h r;
#pragma unroll
    for (int i = 0; i < 8; ++i) r[i] = (_Float16)p[i];
#pragma unroll
    for (int i = 0; i < 8; ++i) r[8 + i] = (_Float16)p[16 + i];
    return r;
}

// ---------------------------------------------------------------------------
// Kernel 1: feats[b,s,:] = hidden[b,s,:] @ W^T + bias  via v_wmma_f32_16x16x32_f16
// M = B*S = 65536 (4096 tiles of 16 rows), K = 768 (24 chunks of 32), N = 20 -> 2 tiles.
// W fragments are staged once per block into LDS, pre-converted to f16 and
// pre-swizzled into WMMA operand layout, so the steady-state loop is just
// A-streaming (global b128) + ds_load_b128 + wmma. Streaming-bound on hidden.
// ---------------------------------------------------------------------------
__global__ __launch_bounds__(256) void feats_gemm_kernel(
    const float* __restrict__ hidden,   // (B*S, H)
    const float* __restrict__ W,        // (T, H)
    const float* __restrict__ bias,     // (T,)
    float* __restrict__ feats)          // (B*S, TPAD)
{
    // B fragments: [kchunk][ntile][lane][16 halves] = 48 KB
    __shared__ _Float16 bfrag[KCHUNKS][2][32][16];

    // ---- cooperative fill: 1536 fragment-lanes, 6 per thread ----
    for (int e = threadIdx.x; e < KCHUNKS * 2 * 32; e += 256) {
        const int kc = e >> 6;          // /64
        const int n  = (e >> 5) & 1;
        const int l  = e & 31;
        int col = n * 16 + (l & 15);
        if (col >= TSZ) col = TSZ - 1;  // clamp pad columns (never stored)
        const int kh = (l >> 4) * 8;
        const float* __restrict__ p = W + (size_t)col * HSZ + kc * 32 + kh;
        _Float16* dst = &bfrag[kc][n][l][0];
#pragma unroll
        for (int i = 0; i < 8; ++i) dst[i] = (_Float16)p[i];
#pragma unroll
        for (int i = 0; i < 8; ++i) dst[8 + i] = (_Float16)p[16 + i];
    }
    __syncthreads();

    const int lane  = threadIdx.x & 31;
    const int wave  = threadIdx.x >> 5;
    const int tile  = blockIdx.x * 8 + wave;     // 0..4095
    const int row0  = tile * 16;
    const int lrow  = lane & 15;
    const int khalf = (lane >> 4) * 8;           // lanes 0-15: K 0..7/16..23; 16-31: 8..15/24..31

    const int col0 = lrow;                        // N tile 0: cols 0..15 (all valid)
    const int col1 = 16 + lrow;                   // N tile 1: cols 16..31 (valid < 20)

    const float* __restrict__ arow = hidden + (size_t)(row0 + lrow) * HSZ + khalf;

    v8f c0 = {};
    v8f c1 = {};

#pragma unroll 4
    for (int kc = 0; kc < KCHUNKS; ++kc) {
        v16h a  = load_frag16(arow + kc * 32);
        v16h b0 = *(const v16h*)&bfrag[kc][0][lane][0];
        v16h b1 = *(const v16h*)&bfrag[kc][1][lane][0];
        c0 = __builtin_amdgcn_wmma_f32_16x16x32_f16(false, a, false, b0,
                                                    (short)0, c0, false, false);
        c1 = __builtin_amdgcn_wmma_f32_16x16x32_f16(false, a, false, b1,
                                                    (short)0, c1, false, false);
    }

    const float bias0 = bias[col0];
    const float bias1 = (col1 < TSZ) ? bias[col1] : 0.0f;
    const int rbase = row0 + ((lane >> 4) * 8);   // C layout: vgpr v -> M = v (+8 for lanes>=16)

#pragma unroll
    for (int v = 0; v < 8; ++v) {
        const size_t r = (size_t)(rbase + v);
        feats[r * TPAD + col0] = c0[v] + bias0;
        if (col1 < TSZ) feats[r * TPAD + col1] = c1[v] + bias1;
    }
}

// ---------------------------------------------------------------------------
// Kernel 2: CRF forward recursion + gold score. One wave (32 lanes) per batch.
// Lane j (< 20) owns tag j; alpha broadcast via lane shuffles; transitions
// column j cached in 20 registers. feats (8 MB) is L2-resident.
// ---------------------------------------------------------------------------
__global__ __launch_bounds__(32) void crf_scan_kernel(
    const float* __restrict__ feats,    // (B*S, TPAD)
    const float* __restrict__ trans,    // (T, T)
    const int*   __restrict__ mask,     // (B, S)
    const int*   __restrict__ labels,   // (B, S)
    float* __restrict__ partial)        // (B,)
{
    const int b = blockIdx.x;
    const int j = threadIdx.x;                 // 0..31
    const bool jv = (j < TSZ);
    const int jc = jv ? j : 0;

    // transitions column j: tcol[i] = trans[i][j]
    float tcol[TSZ];
#pragma unroll
    for (int i = 0; i < TSZ; ++i) tcol[i] = trans[i * TSZ + jc];

    // sequence length (mask is left-contiguous)
    int lsum = 0;
    for (int t = j; t < SSZ; t += 32) lsum += mask[b * SSZ + t];
#pragma unroll
    for (int o = 16; o > 0; o >>= 1) lsum += __shfl_xor(lsum, o, 32);
    const int len = lsum;

    const float NEG = -3.0e38f;
    const size_t fb = (size_t)b * SSZ * TPAD;

    // alpha_0 = feats[b,0,:] + trans[START,:]
    float alpha = jv ? (feats[fb + j] + trans[START_TAG * TSZ + j]) : NEG;

    for (int t = 1; t < len; ++t) {
        const float f = jv ? feats[fb + (size_t)t * TPAD + j] : 0.0f;
        float v[TSZ];
        float m = NEG;
#pragma unroll
        for (int i = 0; i < TSZ; ++i) {
            const float ai = __shfl(alpha, i, 32);
            v[i] = ai + tcol[i];
            m = fmaxf(m, v[i]);
        }
        float s = 0.0f;
#pragma unroll
        for (int i = 0; i < TSZ; ++i) s += __expf(v[i] - m);
        const float na = m + __logf(s) + f;
        alpha = jv ? na : NEG;
    }

    // forward_score_b = logsumexp_j(alpha_j + trans[j][STOP])
    float fv = jv ? (alpha + trans[j * TSZ + STOP_TAG]) : NEG;
    float m = fv;
#pragma unroll
    for (int o = 16; o > 0; o >>= 1) m = fmaxf(m, __shfl_xor(m, o, 32));
    float e = __expf(fv - m);   // pad lanes contribute exp(-huge) = 0
#pragma unroll
    for (int o = 16; o > 0; o >>= 1) e += __shfl_xor(e, o, 32);
    const float fscore = m + __logf(e);

    // gold score: sum_t [trans[prev,cur] + feats[b,t,cur]] + trans[end,STOP]
    float g = 0.0f;
    for (int t = j; t < len; t += 32) {
        const int cur  = labels[b * SSZ + t];
        const int prev = (t == 0) ? START_TAG : labels[b * SSZ + t - 1];
        g += trans[prev * TSZ + cur] + feats[fb + (size_t)t * TPAD + cur];
        if (t == len - 1) g += trans[cur * TSZ + STOP_TAG];
    }
#pragma unroll
    for (int o = 16; o > 0; o >>= 1) g += __shfl_xor(g, o, 32);

    if (j == 0) partial[b] = fscore - g;
}

// ---------------------------------------------------------------------------
// Kernel 3: deterministic reduction of 128 per-batch partials to scalar loss.
// ---------------------------------------------------------------------------
__global__ __launch_bounds__(128) void final_reduce_kernel(
    const float* __restrict__ partial, float* __restrict__ out)
{
    __shared__ float sm[128];
    sm[threadIdx.x] = partial[threadIdx.x];
    __syncthreads();
#pragma unroll
    for (int s = 64; s > 0; s >>= 1) {
        if (threadIdx.x < s) sm[threadIdx.x] += sm[threadIdx.x + s];
        __syncthreads();
    }
    if (threadIdx.x == 0) out[0] = sm[0];
}

extern "C" void kernel_launch(void* const* d_in, const int* in_sizes, int n_in,
                              void* d_out, int out_size, void* d_ws, size_t ws_size,
                              hipStream_t stream) {
    const float* hidden = (const float*)d_in[0];   // (B,S,H) f32
    const float* W      = (const float*)d_in[1];   // (T,H)   f32
    const float* bias   = (const float*)d_in[2];   // (T,)    f32
    const float* trans  = (const float*)d_in[3];   // (T,T)   f32
    const int*   mask   = (const int*)d_in[4];     // (B,S)   i32
    const int*   labels = (const int*)d_in[5];     // (B,S)   i32

    float* feats   = (float*)d_ws;                              // B*S*TPAD floats (8 MB)
    float* partial = feats + (size_t)BSZ * SSZ * TPAD;          // B floats

    // 65536 rows / 16 per tile = 4096 tiles; 8 waves (tiles) per 256-thread block
    feats_gemm_kernel<<<(BSZ * SSZ / 16) / 8, 256, 0, stream>>>(hidden, W, bias, feats);
    crf_scan_kernel<<<BSZ, 32, 0, stream>>>(feats, trans, mask, labels, partial);
    final_reduce_kernel<<<1, 128, 0, stream>>>(partial, (float*)d_out);
}
